// EnhancedMambaFusion_29248727286524
// MI455X (gfx1250) — matrix operand order repaired and checked by reading, not compiled, not run on hardware
//
#include <hip/hip_runtime.h>
#include <hip/hip_bf16.h>
#include <math.h>
#include <stdint.h>

// ---------------------------------------------------------------------------
// Problem constants (from reference)
// ---------------------------------------------------------------------------
#define BATCH   8192
#define DIM     1024
#define D_MODEL 2048
#define D_INNER 4096
#define D_STATE 16
#define DT_RANK 128
#define OUTDIM  256

// ---------------------------------------------------------------------------
// Vector types for WMMA (CDNA5 / gfx1250, wave32)
// ---------------------------------------------------------------------------
typedef _Float16 v16h __attribute__((ext_vector_type(16)));
typedef _Float16 v8h  __attribute__((ext_vector_type(8)));
typedef float    v8f  __attribute__((ext_vector_type(8)));
typedef float    v4f  __attribute__((ext_vector_type(4)));

// GEMM tiling: 128x128 block tile, K-step 32 (v_wmma_f32_16x16x32_f16)
#define BM 128
#define BN 128
#define BK 32
#define LDSH 40                 // halves per LDS row = 80 B (16B aligned, skewed)
#define BUF_HALVES (2 * BM * LDSH)   // A+B tile per buffer (halves)
#define BUF_BYTES  (BUF_HALVES * 2)

// Epilogues
#define EPI_NONE     0
#define EPI_RELU     1
#define EPI_SOFTPLUS 2

// ---------------------------------------------------------------------------
// CDNA5 async copy global->LDS (ASYNCcnt-tracked), 16 bytes per lane.
// ---------------------------------------------------------------------------
__device__ __forceinline__ void async_copy_b128(uint32_t lds_byte_off, uint64_t gaddr)
{
  asm volatile("global_load_async_to_lds_b128 %0, %1, off"
               :: "v"(lds_byte_off), "v"(gaddr)
               : "memory");
}
__device__ __forceinline__ void wait_async0()
{
  asm volatile("s_wait_asynccnt 0x0" ::: "memory");
}

// ---------------------------------------------------------------------------
// f16 WMMA GEMM:  C[M,N] = epi( A[M,K] @ W[N,K]^T + bias[N] )
// A,W row-major f16 (K-contiguous).  Optional f32 output C and/or f16 output Ch.
// M % 128 == 0; K % 32 == 0; N ragged OK.
// Double-buffered LDS fed by GLOBAL_LOAD_ASYNC_TO_LDS_B128.
// ---------------------------------------------------------------------------
template <int EPI>
__global__ __launch_bounds__(256)
void gemm_wmma_f16(const _Float16* __restrict__ A, long lda,
                   const _Float16* __restrict__ W, long ldw,
                   const float* __restrict__ bias,
                   float* __restrict__ C, long ldc,
                   _Float16* __restrict__ Ch, long ldch,
                   int M, int N, int K)
{
  __shared__ _Float16 smem[2 * BUF_HALVES];   // [buf][A|B][128][LDSH]  = 40 KB

  const int tid   = threadIdx.x;
  const int lane  = tid & 31;
  const int wav   = tid >> 5;          // 0..7
  const int wm    = (wav & 3) * 32;    // wave M base inside tile
  const int wn    = (wav >> 2) * 64;   // wave N base inside tile
  const int lmod  = lane & 15;
  const int lhalf = lane >> 4;

  const long bm = (long)blockIdx.y * BM;
  const long bn = (long)blockIdx.x * BN;

  v8f acc[2][4] = {};

  // staging assignment: thread t owns (row = t>>1, half = t&1) of each tile
  const int  srow = tid >> 1;          // 0..127
  const int  skh  = (tid & 1) << 4;    // 0 or 16 halves
  const long arow = bm + srow;         // always < M
  const long brow = bn + srow;
  const bool bok  = (brow < (long)N);

  const _Float16* agp = A + arow * lda + skh;
  const _Float16* wgp = bok ? (W + brow * ldw + skh) : A;

  const uint32_t smem_base = (uint32_t)(uintptr_t)&smem[0];
  const uint32_t a_lds = smem_base + (uint32_t)((srow * LDSH + skh) * 2);
  const uint32_t b_lds = a_lds + (uint32_t)(BM * LDSH * 2);

  // zero-fill OOB W rows once (async never writes them; zeros persist)
  if (!bok) {
    v8h z;
#pragma unroll
    for (int j = 0; j < 8; ++j) z[j] = (_Float16)0.f;
#pragma unroll
    for (int b = 0; b < 2; ++b) {
      _Float16* p = smem + b * BUF_HALVES + BM * LDSH + srow * LDSH + skh;
      *(v8h*)p = z;
      *(v8h*)(p + 8) = z;
    }
  }

  // issue async copies of one K-tile into buffer b
  auto issue_tile = [&](int b, long k0) {
    const uint64_t ga = (uint64_t)(uintptr_t)(agp + k0);
    const uint32_t la = a_lds + (uint32_t)b * BUF_BYTES;
    async_copy_b128(la,      ga);
    async_copy_b128(la + 16, ga + 16);
    if (bok) {
      const uint64_t gb = (uint64_t)(uintptr_t)(wgp + k0);
      const uint32_t lb = b_lds + (uint32_t)b * BUF_BYTES;
      async_copy_b128(lb,      gb);
      async_copy_b128(lb + 16, gb + 16);
    }
  };

  // compute one K-tile from buffer b
  auto compute_tile = [&](int b) {
    const _Float16* Asb = smem + b * BUF_HALVES;
    const _Float16* Bsb = Asb + BM * LDSH;
    v16h bfrag[4];
#pragma unroll
    for (int ni = 0; ni < 4; ++ni) {
      const _Float16* bp = &Bsb[(wn + ni * 16 + lmod) * LDSH + lhalf * 16];
      v8h lo = *(const v8h*)bp;
      v8h hi = *(const v8h*)(bp + 8);
      bfrag[ni] = __builtin_shufflevector(lo, hi,
                    0, 1, 2, 3, 4, 5, 6, 7, 8, 9, 10, 11, 12, 13, 14, 15);
    }
#pragma unroll
    for (int mi = 0; mi < 2; ++mi) {
      const _Float16* ap = &Asb[(wm + mi * 16 + lmod) * LDSH];
      v8h lo = *(const v8h*)(ap + lhalf * 16);
      v8h hi = *(const v8h*)(ap + lhalf * 16 + 8);
      v16h afrag = __builtin_shufflevector(lo, hi,
                    0, 1, 2, 3, 4, 5, 6, 7, 8, 9, 10, 11, 12, 13, 14, 15);
#pragma unroll
      for (int ni = 0; ni < 4; ++ni) {
        acc[mi][ni] = __builtin_amdgcn_wmma_f32_16x16x32_f16(
            false, afrag, false, bfrag[ni],
            (short)0, acc[mi][ni], false, false);
      }
    }
  };

  // ---- software pipeline: tile k+1 streams to LDS while tile k is in WMMA ----
  issue_tile(0, 0);
  wait_async0();
  __syncthreads();

  int cur = 0;
  for (long k0 = 0; k0 < K; k0 += BK) {
    if (k0 + BK < K) issue_tile(cur ^ 1, k0 + BK);
    compute_tile(cur);
    wait_async0();        // my async writes for next tile landed
    __syncthreads();      // everyone done reading cur + next tile visible
    cur ^= 1;
  }

  // ---- epilogue ----
#pragma unroll
  for (int mi = 0; mi < 2; ++mi) {
#pragma unroll
    for (int ni = 0; ni < 4; ++ni) {
      long col = bn + wn + ni * 16 + lmod;
      if (col >= (long)N) continue;
      float bv = bias ? bias[col] : 0.0f;
      long rbase = bm + wm + mi * 16 + (lhalf ? 8 : 0);
#pragma unroll
      for (int j = 0; j < 8; ++j) {
        float v = acc[mi][ni][j] + bv;
        if (EPI == EPI_RELU)     v = v > 0.f ? v : 0.f;
        if (EPI == EPI_SOFTPLUS) v = (v > 20.f) ? v : log1pf(expf(v));
        if (C)  C[(rbase + j) * ldc + col]  = v;
        if (Ch) Ch[(rbase + j) * ldch + col] = (_Float16)v;
      }
    }
  }
}

// ---------------------------------------------------------------------------
// f32 -> f16 bulk convert (weights / input features), 8 elems per thread
// ---------------------------------------------------------------------------
__global__ __launch_bounds__(256)
void cvt16_kernel(const float* __restrict__ src, _Float16* __restrict__ dst, long n)
{
  long i = ((long)blockIdx.x * 256 + threadIdx.x) * 8;
  if (i + 8 <= n) {
    v4f f0 = *(const v4f*)(src + i);
    v4f f1 = *(const v4f*)(src + i + 4);
    v8h h;
#pragma unroll
    for (int j = 0; j < 4; ++j) { h[j] = (_Float16)f0[j]; h[4 + j] = (_Float16)f1[j]; }
    *(v8h*)(dst + i) = h;
  } else {
    for (; i < n; ++i) dst[i] = (_Float16)src[i];
  }
}

// ---------------------------------------------------------------------------
// LayerNorm (+optional residual); writes f32 and optional f16 copy.
// ---------------------------------------------------------------------------
__global__ __launch_bounds__(256)
void ln_kernel(const float* __restrict__ x, long ldx,
               const float* __restrict__ res, long ldres,
               const float* __restrict__ g, const float* __restrict__ b,
               float* __restrict__ out, long ldout,
               _Float16* __restrict__ out16, long ldo16, int n)
{
  __shared__ float s1[256];
  __shared__ float s2[256];
  const long row = blockIdx.x;
  const float* xr = x + row * ldx;
  const float* rr = res ? (res + row * ldres) : nullptr;

  float s = 0.f, q = 0.f;
  for (int j = threadIdx.x; j < n; j += 256) {
    float v = xr[j] + (rr ? rr[j] : 0.f);
    s += v; q += v * v;
  }
  s1[threadIdx.x] = s; s2[threadIdx.x] = q;
  __syncthreads();
  for (int off = 128; off > 0; off >>= 1) {
    if (threadIdx.x < off) {
      s1[threadIdx.x] += s1[threadIdx.x + off];
      s2[threadIdx.x] += s2[threadIdx.x + off];
    }
    __syncthreads();
  }
  const float mean = s1[0] / (float)n;
  const float var  = s2[0] / (float)n - mean * mean;
  const float rstd = rsqrtf(var + 1e-5f);

  float* orow = out + row * ldout;
  _Float16* o16 = out16 ? (out16 + row * ldo16) : nullptr;
  for (int j = threadIdx.x; j < n; j += 256) {
    float v = xr[j] + (rr ? rr[j] : 0.f);
    float r = (v - mean) * rstd * g[j] + b[j];
    orow[j] = r;
    if (o16) o16[j] = (_Float16)r;
  }
}

// ---------------------------------------------------------------------------
// x = silu(x * conv_w[:, -1] + conv_b); writes f32 back into xz and f16 to x16
// ---------------------------------------------------------------------------
__global__ __launch_bounds__(256)
void conv_silu_kernel(float* __restrict__ xz,
                      const float* __restrict__ cw,   // (D_INNER,4) -> [.,3]
                      const float* __restrict__ cb,
                      _Float16* __restrict__ x16)
{
  long i = (long)blockIdx.x * 256 + threadIdx.x;
  if (i >= (long)BATCH * D_INNER) return;
  long row = i >> 12;
  int  col = (int)(i & 4095);
  float* p = xz + row * (2 * (long)D_INNER) + col;
  float v = *p * cw[col * 4 + 3] + cb[col];
  float sv = v / (1.f + expf(-v));
  *p = sv;
  x16[row * (long)D_INNER + col] = (_Float16)sv;
}

// ---------------------------------------------------------------------------
// y = (dt*x*bc + D*x) * silu(z)  -> f16 into y16 (reuses x16 buffer)
// ---------------------------------------------------------------------------
__global__ __launch_bounds__(256)
void mamba_merge_kernel(const float* __restrict__ xz,
                        const float* __restrict__ dt,
                        const float* __restrict__ dbl,
                        const float* __restrict__ Dv,
                        _Float16* __restrict__ y16)
{
  __shared__ float sred[D_STATE];
  __shared__ float sbc;
  const long row = blockIdx.x;
  const float* dr = dbl + row * (DT_RANK + 2 * D_STATE);
  if (threadIdx.x < D_STATE)
    sred[threadIdx.x] = dr[DT_RANK + threadIdx.x] * dr[DT_RANK + D_STATE + threadIdx.x];
  __syncthreads();
  if (threadIdx.x == 0) {
    float s = 0.f;
#pragma unroll
    for (int i = 0; i < D_STATE; ++i) s += sred[i];
    sbc = s;
  }
  __syncthreads();
  const float bc = sbc;

  const float* xr = xz + row * (2 * (long)D_INNER);
  const float* dtr = dt + row * (long)D_INNER;
  _Float16* yr = y16 + row * (long)D_INNER;
  for (int j = threadIdx.x; j < D_INNER; j += 256) {
    float xv = xr[j];
    float zv = xr[D_INNER + j];
    float sz = zv / (1.f + expf(-zv));
    yr[j] = (_Float16)((dtr[j] * xv * bc + Dv[j] * xv) * sz);
  }
}

// ---------------------------------------------------------------------------
// gate logits (N=2) + softmax. One block per row.
// ---------------------------------------------------------------------------
__global__ __launch_bounds__(256)
void gate_softmax_kernel(const float* __restrict__ gh,   // (B,1024)
                         const float* __restrict__ w,    // (2,1024)
                         const float* __restrict__ b,    // (2,)
                         float* __restrict__ out)        // (B,2)
{
  __shared__ float s0[256];
  __shared__ float s1[256];
  const long row = blockIdx.x;
  const float* xr = gh + row * (long)DIM;
  float a0 = 0.f, a1 = 0.f;
  for (int j = threadIdx.x; j < DIM; j += 256) {
    float v = xr[j];
    a0 += v * w[j];
    a1 += v * w[DIM + j];
  }
  s0[threadIdx.x] = a0; s1[threadIdx.x] = a1;
  __syncthreads();
  for (int off = 128; off > 0; off >>= 1) {
    if (threadIdx.x < off) {
      s0[threadIdx.x] += s0[threadIdx.x + off];
      s1[threadIdx.x] += s1[threadIdx.x + off];
    }
    __syncthreads();
  }
  if (threadIdx.x == 0) {
    float l0 = s0[0] + b[0];
    float l1 = s1[0] + b[1];
    float m  = l0 > l1 ? l0 : l1;
    float e0 = expf(l0 - m), e1 = expf(l1 - m);
    float inv = 1.f / (e0 + e1);
    out[row * 2 + 0] = e0 * inv;
    out[row * 2 + 1] = e1 * inv;
  }
}

// ---------------------------------------------------------------------------
// host-side helpers
// ---------------------------------------------------------------------------
static inline void run_gemm(int epi,
                            const _Float16* A, long lda,
                            const _Float16* W, long ldw,
                            const float* bias,
                            float* C, long ldc,
                            _Float16* Ch, long ldch,
                            int M, int N, int K, hipStream_t s)
{
  dim3 grid((unsigned)((N + BN - 1) / BN), (unsigned)((M + BM - 1) / BM));
  dim3 block(256);
  if (epi == EPI_RELU)
    gemm_wmma_f16<EPI_RELU><<<grid, block, 0, s>>>(A, lda, W, ldw, bias, C, ldc, Ch, ldch, M, N, K);
  else if (epi == EPI_SOFTPLUS)
    gemm_wmma_f16<EPI_SOFTPLUS><<<grid, block, 0, s>>>(A, lda, W, ldw, bias, C, ldc, Ch, ldch, M, N, K);
  else
    gemm_wmma_f16<EPI_NONE><<<grid, block, 0, s>>>(A, lda, W, ldw, bias, C, ldc, Ch, ldch, M, N, K);
}

static inline void run_cvt(const float* s, _Float16* d, long n, hipStream_t st)
{
  long blocks = (n / 8 + 255) / 256;
  cvt16_kernel<<<(unsigned)blocks, 256, 0, st>>>(s, d, n);
}

extern "C" void kernel_launch(void* const* d_in, const int* in_sizes, int n_in,
                              void* d_out, int out_size, void* d_ws, size_t ws_size,
                              hipStream_t stream)
{
  // ---- inputs (setup_inputs order) ----
  const float* image_features = (const float*)d_in[0];
  const float* text_features  = (const float*)d_in[1];
  const float* img_w    = (const float*)d_in[2];
  const float* img_b    = (const float*)d_in[3];
  const float* img_g    = (const float*)d_in[4];
  const float* img_beta = (const float*)d_in[5];
  const float* txt_w    = (const float*)d_in[6];
  const float* txt_b    = (const float*)d_in[7];
  const float* txt_g    = (const float*)d_in[8];
  const float* txt_beta = (const float*)d_in[9];
  const float* m_in_w   = (const float*)d_in[10];
  const float* m_conv_w = (const float*)d_in[11];
  const float* m_conv_b = (const float*)d_in[12];
  const float* m_x_w    = (const float*)d_in[13];
  const float* m_dt_w   = (const float*)d_in[14];
  const float* m_dt_b   = (const float*)d_in[15];
  // d_in[16] = m_Alog (unused in forward)
  const float* m_D      = (const float*)d_in[17];
  const float* m_out_w  = (const float*)d_in[18];
  const float* m_ln_g   = (const float*)d_in[19];
  const float* m_ln_b   = (const float*)d_in[20];
  const float* attn_in_w  = (const float*)d_in[21];
  const float* attn_in_b  = (const float*)d_in[22];
  const float* attn_out_w = (const float*)d_in[23];
  const float* attn_out_b = (const float*)d_in[24];
  const float* gate1_w  = (const float*)d_in[25];
  const float* gate1_b  = (const float*)d_in[26];
  const float* gate2_w  = (const float*)d_in[27];
  const float* gate2_b  = (const float*)d_in[28];
  const float* fc_w     = (const float*)d_in[29];
  const float* fc_b     = (const float*)d_in[30];
  const float* fin_g    = (const float*)d_in[31];
  const float* fin_b    = (const float*)d_in[32];

  float* out_fused = (float*)d_out;                        // (B, OUTDIM)
  float* out_gate  = (float*)d_out + (long)BATCH * OUTDIM; // (B, 2)

  const int B = BATCH;

  // ---- workspace carve-out: f32 region then f16 region ----
  float* f32p = (float*)d_ws;
  size_t o32 = 0;
  auto carve32 = [&](size_t n) { float* p = f32p + o32; o32 += (n + 63) & ~(size_t)63; return p; };
  float* h_buf  = carve32((size_t)B * D_MODEL);
  float* tA     = carve32((size_t)B * D_MODEL);
  float* tB     = carve32((size_t)B * D_MODEL);
  float* xz     = carve32((size_t)B * 2 * D_INNER);
  float* dt_buf = carve32((size_t)B * D_INNER);
  float* dbl    = carve32((size_t)B * (DT_RANK + 2 * D_STATE));

  _Float16* f16p = (_Float16*)(f32p + o32);
  size_t o16 = 0;
  auto carve16 = [&](size_t n) { _Float16* p = f16p + o16; o16 += (n + 63) & ~(size_t)63; return p; };
  // f16 activations
  _Float16* img16 = carve16((size_t)B * DIM);
  _Float16* txt16 = carve16((size_t)B * DIM);
  _Float16* h16   = carve16((size_t)B * D_MODEL);
  _Float16* x16   = carve16((size_t)B * D_INNER);    // x, later y
  _Float16* dbl16 = carve16((size_t)B * (DT_RANK + 2 * D_STATE));
  _Float16* v16   = carve16((size_t)B * D_MODEL);
  _Float16* a16   = carve16((size_t)B * D_MODEL);
  // f16 weights
  _Float16* w_img   = carve16((size_t)DIM * DIM);
  _Float16* w_txt   = carve16((size_t)DIM * DIM);
  _Float16* w_in    = carve16((size_t)2 * 2 * D_INNER * D_MODEL);
  _Float16* w_x     = carve16((size_t)2 * (DT_RANK + 2 * D_STATE) * D_INNER);
  _Float16* w_dt    = carve16((size_t)2 * D_INNER * DT_RANK);
  _Float16* w_out   = carve16((size_t)2 * D_MODEL * D_INNER);
  _Float16* w_attnv = carve16((size_t)D_MODEL * D_MODEL);
  _Float16* w_attno = carve16((size_t)D_MODEL * D_MODEL);
  _Float16* w_g1    = carve16((size_t)DIM * D_MODEL);
  _Float16* w_fc    = carve16((size_t)OUTDIM * D_MODEL);

  // ---- one-time f32->f16 conversions (weights stream from L2 afterwards) ----
  run_cvt(image_features, img16, (long)B * DIM, stream);
  run_cvt(text_features,  txt16, (long)B * DIM, stream);
  run_cvt(img_w,   w_img, (long)DIM * DIM, stream);
  run_cvt(txt_w,   w_txt, (long)DIM * DIM, stream);
  run_cvt(m_in_w,  w_in,  (long)2 * 2 * D_INNER * D_MODEL, stream);
  run_cvt(m_x_w,   w_x,   (long)2 * (DT_RANK + 2 * D_STATE) * D_INNER, stream);
  run_cvt(m_dt_w,  w_dt,  (long)2 * D_INNER * DT_RANK, stream);
  run_cvt(m_out_w, w_out, (long)2 * D_MODEL * D_INNER, stream);
  run_cvt(attn_in_w + (size_t)2 * D_MODEL * D_MODEL, w_attnv, (long)D_MODEL * D_MODEL, stream);
  run_cvt(attn_out_w, w_attno, (long)D_MODEL * D_MODEL, stream);
  run_cvt(gate1_w, w_g1, (long)DIM * D_MODEL, stream);
  run_cvt(fc_w,    w_fc, (long)OUTDIM * D_MODEL, stream);

  // ---- input projections + LN -> h (f32 + f16) ----
  run_gemm(EPI_RELU, img16, DIM, w_img, DIM, img_b, tA, DIM, nullptr, 0, B, DIM, DIM, stream);
  ln_kernel<<<B, 256, 0, stream>>>(tA, DIM, nullptr, 0, img_g, img_beta,
                                   h_buf, D_MODEL, h16, D_MODEL, DIM);
  run_gemm(EPI_RELU, txt16, DIM, w_txt, DIM, txt_b, tB, DIM, nullptr, 0, B, DIM, DIM, stream);
  ln_kernel<<<B, 256, 0, stream>>>(tB, DIM, nullptr, 0, txt_g, txt_beta,
                                   h_buf + DIM, D_MODEL, h16 + DIM, D_MODEL, DIM);

  // ---- 2 mamba layers ----
  for (int l = 0; l < 2; ++l) {
    const _Float16* in_w = w_in  + (size_t)l * 2 * D_INNER * D_MODEL;
    const float*    cw   = m_conv_w + (size_t)l * D_INNER * 4;
    const float*    cb   = m_conv_b + (size_t)l * D_INNER;
    const _Float16* xw   = w_x   + (size_t)l * (DT_RANK + 2 * D_STATE) * D_INNER;
    const _Float16* dtw  = w_dt  + (size_t)l * D_INNER * DT_RANK;
    const float*    dtb  = m_dt_b + (size_t)l * D_INNER;
    const float*    Dv   = m_D   + (size_t)l * D_INNER;
    const _Float16* ow   = w_out + (size_t)l * D_MODEL * D_INNER;
    const float*    lng  = m_ln_g + (size_t)l * D_MODEL;
    const float*    lnb  = m_ln_b + (size_t)l * D_MODEL;

    // xz = h @ in_w^T  (B x 8192, f32)
    run_gemm(EPI_NONE, h16, D_MODEL, in_w, D_MODEL, nullptr,
             xz, 2 * D_INNER, nullptr, 0, B, 2 * D_INNER, D_MODEL, stream);
    // x = silu(x*cw+cb) -> xz (f32) and x16 (f16)
    {
      long total = (long)B * D_INNER;
      conv_silu_kernel<<<(unsigned)((total + 255) / 256), 256, 0, stream>>>(xz, cw, cb, x16);
    }
    // dbl = x @ xw^T  (B x 160, f32 + f16)
    run_gemm(EPI_NONE, x16, D_INNER, xw, D_INNER, nullptr,
             dbl, DT_RANK + 2 * D_STATE, dbl16, DT_RANK + 2 * D_STATE,
             B, DT_RANK + 2 * D_STATE, D_INNER, stream);
    // dt = softplus(dbl[:, :128] @ dtw^T + dtb)  (B x 4096, f32)
    run_gemm(EPI_SOFTPLUS, dbl16, DT_RANK + 2 * D_STATE, dtw, DT_RANK, dtb,
             dt_buf, D_INNER, nullptr, 0, B, D_INNER, DT_RANK, stream);
    // y = (dt*x*bc + D*x)*silu(z) -> x16 (f16)
    mamba_merge_kernel<<<B, 256, 0, stream>>>(xz, dt_buf, dbl, Dv, x16);
    // y2 = y @ ow^T  (B x 2048, f32)
    run_gemm(EPI_NONE, x16, D_INNER, ow, D_INNER, nullptr,
             tA, D_MODEL, nullptr, 0, B, D_MODEL, D_INNER, stream);
    // h = LN(y2 + h) (f32 + f16)
    ln_kernel<<<B, 256, 0, stream>>>(tA, D_MODEL, h_buf, D_MODEL, lng, lnb,
                                     h_buf, D_MODEL, h16, D_MODEL, D_MODEL);
  }

  // ---- tail ----
  // v = h @ attn_in_w[2*D_MODEL:]^T + b   (f16 only)
  run_gemm(EPI_NONE, h16, D_MODEL, w_attnv, D_MODEL, attn_in_b + 2 * D_MODEL,
           nullptr, 0, v16, D_MODEL, B, D_MODEL, D_MODEL, stream);
  // attn_out = v @ attn_out_w^T + b       (f16 only)
  run_gemm(EPI_NONE, v16, D_MODEL, w_attno, D_MODEL, attn_out_b,
           nullptr, 0, a16, D_MODEL, B, D_MODEL, D_MODEL, stream);
  // gate_h = relu(attn_out @ gate1_w^T + b)  (B x 1024, f32)
  run_gemm(EPI_RELU, a16, D_MODEL, w_g1, D_MODEL, gate1_b,
           tA, DIM, nullptr, 0, B, DIM, D_MODEL, stream);
  // gate_weights = softmax(gate_h @ gate2_w^T + b)
  gate_softmax_kernel<<<B, 256, 0, stream>>>(tA, gate2_w, gate2_b, out_gate);

  // fused = LN(h @ fc_w^T + fc_b)  (B x 256)
  run_gemm(EPI_NONE, h16, D_MODEL, w_fc, D_MODEL, fc_b,
           tB, OUTDIM, nullptr, 0, B, OUTDIM, D_MODEL, stream);
  ln_kernel<<<B, 256, 0, stream>>>(tB, OUTDIM, nullptr, 0, fin_g, fin_b,
                                   out_fused, OUTDIM, nullptr, 0, OUTDIM);
}
